// MyKANLayer_14035953123870
// MI455X (gfx1250) — matrix operand order repaired and checked by reading, not compiled, not run on hardware
//
#include <hip/hip_runtime.h>
#include <hip/hip_bf16.h>

typedef __attribute__((ext_vector_type(16))) _Float16 v16h;
typedef __attribute__((ext_vector_type(8)))  _Float16 v8h;
typedef __attribute__((ext_vector_type(2)))  _Float16 v2h;
typedef __attribute__((ext_vector_type(8)))  float    v8f;

// K layout: [0,128)=x^1  [128,256)=x^2  [256,384)=x^3  [384,512)=ones (bias via coeff d=0)
#define KTOT 512
#define ROWH 520   // LDS row stride in halfs: 512 + 8 pad -> 260 dwords -> conflict-free

__global__ __launch_bounds__(256)
void kan_wmma_f16(const float* __restrict__ x,
                  const float* __restrict__ coeff,
                  float* __restrict__ out)
{
    __shared__ _Float16 Wl[128 * ROWH];   // 133,120 B (CDNA5 LDS: up to 320 KB/WG)

    const int t = threadIdx.x;

    // ---------------- Phase 1: stage W into LDS as f16, B-operand friendly ----------------
    // thread t handles j = (t&63)*2, j+1 ; o = i*4 + (t>>6)  (coalesced float4 loads)
    {
        const int jj = (t & 63) * 2;
        const int ob = t >> 6;
        const float4* c4 = (const float4*)coeff;   // coeff[o*128+j][0..3]
        #pragma unroll 4
        for (int i = 0; i < 32; ++i) {
            const int o = i * 4 + ob;
            const float4 ca = c4[o * 128 + jj];
            const float4 cb = c4[o * 128 + jj + 1];
            v2h* dst = (v2h*)(Wl + o * ROWH);
            const int jh = jj >> 1;
            v2h p;
            p.x = (_Float16)ca.y; p.y = (_Float16)cb.y; dst[jh]       = p; // d=1 -> K sec 0
            p.x = (_Float16)ca.z; p.y = (_Float16)cb.z; dst[ 64 + jh] = p; // d=2 -> K sec 1
            p.x = (_Float16)ca.w; p.y = (_Float16)cb.w; dst[128 + jh] = p; // d=3 -> K sec 2
            p.x = (_Float16)ca.x; p.y = (_Float16)cb.x; dst[192 + jh] = p; // d=0 -> K sec 3 (bias)
        }
    }
    __syncthreads();

    // ---------------- Phase 2: WMMA GEMM  out = P @ W^T ----------------
    const int lane  = t & 31;
    const int wv    = t >> 5;      // 0..7
    const int r     = lane & 15;   // A row-in-tile / B col-in-tile / D col
    const int kh    = lane >> 4;   // K-half selector
    const int mtile = wv & 3;      // 4 M-tiles per WG
    const int nhalf = wv >> 2;     // 2 N-halves per WG

    const int mbase = blockIdx.x * 64 + mtile * 16;
    const float* xr = x + (size_t)(mbase + r) * 128;

    // Build A fragments (k order used here and in B: e0..15 -> K kh*16..kh*16+15 contiguous)
    v16h A[12];
    #pragma unroll
    for (int blk = 0; blk < 4; ++blk) {
        const int j0 = blk * 32 + kh * 16;
        const float4 f0 = *(const float4*)(xr + j0);
        const float4 f1 = *(const float4*)(xr + j0 + 4);
        const float4 f2 = *(const float4*)(xr + j0 + 8);
        const float4 f3 = *(const float4*)(xr + j0 + 12);
        float f[16] = { f0.x,f0.y,f0.z,f0.w, f1.x,f1.y,f1.z,f1.w,
                        f2.x,f2.y,f2.z,f2.w, f3.x,f3.y,f3.z,f3.w };
        #pragma unroll
        for (int e = 0; e < 16; ++e) {
            const float v1 = f[e];
            const float v2 = v1 * v1;
            const float v3 = v2 * v1;
            A[blk    ][e] = (_Float16)v1;
            A[4 + blk][e] = (_Float16)v2;
            A[8 + blk][e] = (_Float16)v3;
        }
    }
    v16h AO;
    #pragma unroll
    for (int e = 0; e < 16; ++e) AO[e] = (_Float16)1.0f;

    #pragma unroll
    for (int nn = 0; nn < 4; ++nn) {
        const int nt = nhalf * 4 + nn;
        // B-operand (32x16 f16): lane holds column o=nt*16+r; lanes 0-15: K kh*16..+15 contiguous
        const _Float16* wr = Wl + (nt * 16 + r) * ROWH + kh * 16;

        // Software-pipelined B-frag loads: prefetch distance 2, 3 rotating buffers,
        // so each WMMA only waits on loads issued two iterations earlier.
        v8h b0[3], b1[3];
        b0[0] = *(const v8h*)(wr);       b1[0] = *(const v8h*)(wr + 8);
        b0[1] = *(const v8h*)(wr + 32);  b1[1] = *(const v8h*)(wr + 40);

        v8f c = {};
        #pragma unroll
        for (int kk = 0; kk < 16; ++kk) {
            const int cur = kk % 3;
            const int nxt = (kk + 2) % 3;
            if (kk < 14) {
                b0[nxt] = *(const v8h*)(wr + (kk + 2) * 32);
                b1[nxt] = *(const v8h*)(wr + (kk + 2) * 32 + 8);
            }
            const v16h B = __builtin_shufflevector(b0[cur], b1[cur],
                0,1,2,3,4,5,6,7,8,9,10,11,12,13,14,15);
            const v16h a = (kk < 12) ? A[kk] : AO;   // ones-section folds in the bias
            c = __builtin_amdgcn_wmma_f32_16x16x32_f16(
                    false, a, false, B, (short)0, c, false, false);
        }

        // D layout: lane -> col n=r ; VGPR v -> row kh*8+v
        float* op = out + (size_t)(mbase + kh * 8) * 128 + nt * 16 + r;
        #pragma unroll
        for (int v = 0; v < 8; ++v)
            op[(size_t)v * 128] = c[v];
    }
}

extern "C" void kernel_launch(void* const* d_in, const int* in_sizes, int n_in,
                              void* d_out, int out_size, void* d_ws, size_t ws_size,
                              hipStream_t stream)
{
    const float* x     = (const float*)d_in[0];   // [B, 128] f32
    const float* coeff = (const float*)d_in[1];   // [128*128, 4] f32
    float* out = (float*)d_out;                   // [B, 128] f32

    const int B = in_sizes[0] / 128;              // 8192
    dim3 grid(B / 64), block(256);
    hipLaunchKernelGGL(kan_wmma_f16, grid, block, 0, stream, x, coeff, out);
}
